// MoE_76192719832095
// MI455X (gfx1250) — compile-verified, matrix operand order checked by reading, not run on hardware
//
#include <hip/hip_runtime.h>
#include <math.h>

typedef __attribute__((ext_vector_type(16))) _Float16 v16h;
typedef __attribute__((ext_vector_type(8)))  _Float16 v8h;
typedef __attribute__((ext_vector_type(4)))  _Float16 v4h;
typedef __attribute__((ext_vector_type(8)))  float    v8f;

static constexpr int Bq = 2048;   // tokens
static constexpr int Hq = 1024;   // hidden
static constexpr int Fq = 4096;   // ffn
static constexpr int Eq = 8;      // experts
static constexpr int TOPK = 2;
static constexpr int NROWS = Bq * TOPK;   // 4096 assigned rows total (always)

static constexpr int BT   = 128;  // block tile (M and N)
static constexpr int KT   = 32;   // k-step (WMMA f16 K)
static constexpr int LDSS = 40;   // padded LDS row stride in halves (80B, 16B aligned)

// -------- CDNA5 async global->LDS staging (guarded; falls back to sync) -----
#if defined(__has_builtin)
#  if __has_builtin(__builtin_amdgcn_global_load_async_to_lds_b128)
#    define HAVE_ASYNC_LDS 1
#  endif
#endif
#ifndef HAVE_ASYNC_LDS
#  define HAVE_ASYNC_LDS 0
#endif

#if HAVE_ASYNC_LDS
typedef int v4i_t __attribute__((vector_size(4 * sizeof(int))));
typedef __attribute__((address_space(1))) v4i_t g_v4i;
typedef __attribute__((address_space(3))) v4i_t l_v4i;
__device__ __forceinline__ void async_cp16(const void* g, void* l) {
  __builtin_amdgcn_global_load_async_to_lds_b128((g_v4i*)g, (l_v4i*)l, 0, 0);
}
__device__ __forceinline__ void wait_async0() {
#if __has_builtin(__builtin_amdgcn_s_wait_asynccnt)
  __builtin_amdgcn_s_wait_asynccnt(0);
#else
  asm volatile("s_wait_asynccnt 0" ::: "memory");
#endif
}
#endif

// ---------------- helpers ----------------

__device__ __forceinline__ float gelu_tanh(float x) {
  float x3 = x * x * x;
  float t  = 0.7978845608028654f * (x + 0.044715f * x3);
  return 0.5f * x * (1.0f + tanhf(t));
}

// Load a 16x32 f16 fragment (A layout) from LDS stored row-major with stride LDSS.
// lane 0-15: row=sub, K 0..7 then 16..23 ; lane 16-31: row=sub, K 8..15 then 24..31.
__device__ __forceinline__ v16h frag_ld(const _Float16* p0) {
  v8h lo = *(const v8h*)(p0);        // K k0..k0+7
  v8h hi = *(const v8h*)(p0 + 16);   // K k0+16..k0+23
  return __builtin_shufflevector(lo, hi, 0,1,2,3,4,5,6,7,8,9,10,11,12,13,14,15);
}

// ---------------- kernels ----------------

__global__ void cvt_x_kernel(const float* __restrict__ x, _Float16* __restrict__ xh) {
  int i = (blockIdx.x * blockDim.x + threadIdx.x) * 4;
  float4 v = *(const float4*)(x + i);
  v4h o = { (_Float16)v.x, (_Float16)v.y, (_Float16)v.z, (_Float16)v.w };
  *(v4h*)(xh + i) = o;
}

__global__ void gate_kernel(const float* __restrict__ x, const float* __restrict__ Wg,
                            const float* __restrict__ alpha,
                            int* __restrict__ topidx, float* __restrict__ topscale,
                            int* __restrict__ counts) {
  int b = blockIdx.x * blockDim.x + threadIdx.x;
  if (b >= Bq) return;
  float acc[Eq];
#pragma unroll
  for (int e = 0; e < Eq; ++e) acc[e] = 0.0f;
  const float* xr = x + (size_t)b * Hq;
  for (int h = 0; h < Hq; ++h) {
    float xv = xr[h];
    const float* wr = Wg + (size_t)h * Eq;
#pragma unroll
    for (int e = 0; e < Eq; ++e) acc[e] += xv * wr[e];
  }
  float m = acc[0];
#pragma unroll
  for (int e = 1; e < Eq; ++e) m = fmaxf(m, acc[e]);
  float p[Eq]; float s = 0.0f;
#pragma unroll
  for (int e = 0; e < Eq; ++e) { p[e] = expf(acc[e] - m); s += p[e]; }
  float inv = 1.0f / s;
  // top-1 (lowest index on tie, matching jax.lax.top_k)
  int i0 = 0; float b0 = p[0];
#pragma unroll
  for (int e = 1; e < Eq; ++e) if (p[e] > b0) { b0 = p[e]; i0 = e; }
  int i1 = -1; float b1 = -1.0f;
#pragma unroll
  for (int e = 0; e < Eq; ++e) if (e != i0 && p[e] > b1) { b1 = p[e]; i1 = e; }
  topidx[2*b + 0] = i0;  topscale[2*b + 0] = b0 * inv * alpha[i0];
  topidx[2*b + 1] = i1;  topscale[2*b + 1] = b1 * inv * alpha[i1];
  atomicAdd(&counts[i0], 1);
  atomicAdd(&counts[i1], 1);
}

__global__ void offsets_kernel(const int* __restrict__ counts,
                               int* __restrict__ offsets, int* __restrict__ cursor) {
  int t = 0;
  for (int e = 0; e < Eq; ++e) { offsets[e] = t; cursor[e] = t; t += counts[e]; }
}

__global__ void dispatch_kernel(const int* __restrict__ topidx, const float* __restrict__ topscale,
                                int* __restrict__ cursor, int* __restrict__ rowmap,
                                float* __restrict__ rowscale, int* __restrict__ tokpos) {
  int b = blockIdx.x * blockDim.x + threadIdx.x;
  if (b >= Bq) return;
#pragma unroll
  for (int k = 0; k < TOPK; ++k) {
    int e = topidx[2*b + k];
    int pos = atomicAdd(&cursor[e], 1);
    rowmap[pos]   = b;
    rowscale[pos] = topscale[2*b + k];
    tokpos[2*b + k] = pos;
  }
}

// GEMM1: h = gelu( gather(xh) @ W1[e] ), f16 A/B, f32 acc, f16 out to hbuf
__global__ __launch_bounds__(256) void gemm1_kernel(
    const _Float16* __restrict__ xh, const float* __restrict__ W1,
    const int* __restrict__ counts, const int* __restrict__ offsets,
    const int* __restrict__ rowmap, _Float16* __restrict__ hbuf) {
  __shared__ _Float16 lA[BT * LDSS];
  __shared__ _Float16 lB[BT * LDSS];
  const int MT = Bq / BT;   // 16 (worst case: one expert owns all tokens)
  const int NT = Fq / BT;   // 32
  int bid = blockIdx.x;
  int e   = bid / (MT * NT);
  int r2  = bid % (MT * NT);
  int mt  = r2 / NT, nt = r2 % NT;
  int rows  = counts[e];
  int mbase = mt * BT;
  if (mbase >= rows) return;                 // block-uniform exit
  int valid = rows - mbase; if (valid > BT) valid = BT;
  int off = offsets[e];
  int n0  = nt * BT;
  const float* W1e = W1 + (size_t)e * Hq * Fq;
  int tid  = threadIdx.x;
  int lane = tid & 31, wave = tid >> 5;
  int wm = wave >> 2, wn = wave & 3;         // 2x4 wave grid, each wave 64x32
  int sub = lane & 15, khalf = lane >> 4;

  // Hoisted A-staging gather (loop-invariant): clamp OOR rows to a valid row;
  // their outputs are never stored, so no zero-fill / divergence needed.
  const _Float16* asrc[2];
  _Float16*       adst[2];
#pragma unroll
  for (int j = 0; j < 2; ++j) {
    int c = tid + j * 256;                   // 512 chunks of 8 halves
    int r = c >> 2, q = c & 3;
    int rr = (r < valid) ? r : (valid - 1);
    int gr = rowmap[off + mbase + rr];
    asrc[j] = xh + (size_t)gr * Hq + q * 8;
    adst[j] = &lA[r * LDSS + q * 8];
  }
  // Hoisted B-staging addresses
  const float* bsrc[4];
  _Float16*    bdst[4];
#pragma unroll
  for (int j = 0; j < 4; ++j) {
    int c  = tid + j * 256;                  // 1024 chunks of 4 f32
    int hh = c >> 5;
    int fq = (c & 31) * 4;
    bsrc[j] = W1e + (size_t)hh * Fq + n0 + fq;
    bdst[j] = &lB[fq * LDSS + hh];
  }

  v8f acc[4][2] = {};
  for (int k0 = 0; k0 < Hq; k0 += KT) {
    // stage A: 128 gathered token rows x 32 halves
#pragma unroll
    for (int j = 0; j < 2; ++j) {
#if HAVE_ASYNC_LDS
      async_cp16(asrc[j] + k0, adst[j]);
#else
      *(v8h*)adst[j] = *(const v8h*)(asrc[j] + k0);
#endif
    }
    // stage B: W1 tile [32 h x 128 f] f32 -> f16, transposed to Bt[f][h]
#pragma unroll
    for (int j = 0; j < 4; ++j) {
      float4 w = *(const float4*)(bsrc[j] + (size_t)k0 * Fq);
      bdst[j][0 * LDSS] = (_Float16)w.x;
      bdst[j][1 * LDSS] = (_Float16)w.y;
      bdst[j][2 * LDSS] = (_Float16)w.z;
      bdst[j][3 * LDSS] = (_Float16)w.w;
    }
    // prefetch next B k-tile (one 64B line per thread covers the 16KB tile)
    if (k0 + KT < Hq) {
      int pr = tid >> 3, pq = (tid & 7) * 16;
      __builtin_prefetch((const void*)(W1e + (size_t)(k0 + KT + pr) * Fq + n0 + pq), 0, 1);
    }
#if HAVE_ASYNC_LDS
    wait_async0();
#endif
    __syncthreads();
    v16h a[4], bf[2];
#pragma unroll
    for (int i = 0; i < 4; ++i)
      a[i] = frag_ld(&lA[(wm * 64 + i * 16 + sub) * LDSS + khalf * 8]);
#pragma unroll
    for (int j = 0; j < 2; ++j)
      bf[j] = frag_ld(&lB[(wn * 32 + j * 16 + sub) * LDSS + khalf * 8]);
#pragma unroll
    for (int i = 0; i < 4; ++i)
#pragma unroll
      for (int j = 0; j < 2; ++j)
        acc[i][j] = __builtin_amdgcn_wmma_f32_16x16x32_f16(
            false, a[i], false, bf[j], (short)0, acc[i][j], false, false);
    __syncthreads();
  }
  // epilogue: gelu -> hbuf (f16)
  int ncol = lane & 15;
  int moff = khalf * 8;
#pragma unroll
  for (int i = 0; i < 4; ++i)
#pragma unroll
    for (int j = 0; j < 2; ++j)
#pragma unroll
      for (int r = 0; r < 8; ++r) {
        int Mloc = wm * 64 + i * 16 + moff + r;
        if (Mloc < valid) {
          int Nloc = wn * 32 + j * 16 + ncol;
          hbuf[(size_t)(off + mbase + Mloc) * Fq + n0 + Nloc] = (_Float16)gelu_tanh(acc[i][j][r]);
        }
      }
}

// GEMM2: eobuf = rowscale * (h @ W2[e]) , f16 A/B, f32 acc/out
__global__ __launch_bounds__(256) void gemm2_kernel(
    const _Float16* __restrict__ hbuf, const float* __restrict__ W2,
    const int* __restrict__ counts, const int* __restrict__ offsets,
    const float* __restrict__ rowscale, float* __restrict__ eobuf) {
  __shared__ _Float16 lA[BT * LDSS];
  __shared__ _Float16 lB[BT * LDSS];
  const int MT = Bq / BT;   // 16
  const int NT = Hq / BT;   // 8
  int bid = blockIdx.x;
  int e   = bid / (MT * NT);
  int r2  = bid % (MT * NT);
  int mt  = r2 / NT, nt = r2 % NT;
  int rows  = counts[e];
  int mbase = mt * BT;
  if (mbase >= rows) return;
  int valid = rows - mbase; if (valid > BT) valid = BT;
  int off = offsets[e];
  int n0  = nt * BT;
  const float* W2e = W2 + (size_t)e * Fq * Hq;
  int tid  = threadIdx.x;
  int lane = tid & 31, wave = tid >> 5;
  int wm = wave >> 2, wn = wave & 3;
  int sub = lane & 15, khalf = lane >> 4;

  const _Float16* asrc[2];
  _Float16*       adst[2];
#pragma unroll
  for (int j = 0; j < 2; ++j) {
    int c = tid + j * 256;
    int r = c >> 2, q = c & 3;
    int rr = (r < valid) ? r : (valid - 1);
    asrc[j] = hbuf + (size_t)(off + mbase + rr) * Fq + q * 8;
    adst[j] = &lA[r * LDSS + q * 8];
  }
  const float* bsrc[4];
  _Float16*    bdst[4];
#pragma unroll
  for (int j = 0; j < 4; ++j) {
    int c  = tid + j * 256;
    int ff = c >> 5;
    int nq = (c & 31) * 4;
    bsrc[j] = W2e + (size_t)ff * Hq + n0 + nq;
    bdst[j] = &lB[nq * LDSS + ff];
  }

  v8f acc[4][2] = {};
  for (int k0 = 0; k0 < Fq; k0 += KT) {
#pragma unroll
    for (int j = 0; j < 2; ++j) {
#if HAVE_ASYNC_LDS
      async_cp16(asrc[j] + k0, adst[j]);
#else
      *(v8h*)adst[j] = *(const v8h*)(asrc[j] + k0);
#endif
    }
#pragma unroll
    for (int j = 0; j < 4; ++j) {
      float4 w = *(const float4*)(bsrc[j] + (size_t)k0 * Hq);
      bdst[j][0 * LDSS] = (_Float16)w.x;
      bdst[j][1 * LDSS] = (_Float16)w.y;
      bdst[j][2 * LDSS] = (_Float16)w.z;
      bdst[j][3 * LDSS] = (_Float16)w.w;
    }
    if (k0 + KT < Fq) {
      int pr = tid >> 3, pq = (tid & 7) * 16;
      __builtin_prefetch((const void*)(W2e + (size_t)(k0 + KT + pr) * Hq + n0 + pq), 0, 1);
    }
#if HAVE_ASYNC_LDS
    wait_async0();
#endif
    __syncthreads();
    v16h a[4], bf[2];
#pragma unroll
    for (int i = 0; i < 4; ++i)
      a[i] = frag_ld(&lA[(wm * 64 + i * 16 + sub) * LDSS + khalf * 8]);
#pragma unroll
    for (int j = 0; j < 2; ++j)
      bf[j] = frag_ld(&lB[(wn * 32 + j * 16 + sub) * LDSS + khalf * 8]);
#pragma unroll
    for (int i = 0; i < 4; ++i)
#pragma unroll
      for (int j = 0; j < 2; ++j)
        acc[i][j] = __builtin_amdgcn_wmma_f32_16x16x32_f16(
            false, a[i], false, bf[j], (short)0, acc[i][j], false, false);
    __syncthreads();
  }
  int ncol = lane & 15;
  int moff = khalf * 8;
#pragma unroll
  for (int i = 0; i < 4; ++i)
#pragma unroll
    for (int j = 0; j < 2; ++j)
#pragma unroll
      for (int r = 0; r < 8; ++r) {
        int Mloc = wm * 64 + i * 16 + moff + r;
        if (Mloc < valid) {
          int pos  = off + mbase + Mloc;
          int Nloc = wn * 32 + j * 16 + ncol;
          eobuf[(size_t)pos * Hq + n0 + Nloc] = rowscale[pos] * acc[i][j][r];
        }
      }
}

__global__ void combine_kernel(const float* __restrict__ eobuf, const int* __restrict__ tokpos,
                               float* __restrict__ out) {
  int i4 = blockIdx.x * blockDim.x + threadIdx.x;   // one float4 per thread
  int b  = i4 / (Hq / 4);
  int q  = i4 % (Hq / 4);
  int p0 = tokpos[2*b], p1 = tokpos[2*b + 1];
  const float4* e4 = (const float4*)eobuf;
  float4 a = e4[(size_t)p0 * (Hq / 4) + q];
  float4 c = e4[(size_t)p1 * (Hq / 4) + q];
  float4 o; o.x = a.x + c.x; o.y = a.y + c.y; o.z = a.z + c.z; o.w = a.w + c.w;
  ((float4*)out)[i4] = o;
}

// ---------------- launcher ----------------

extern "C" void kernel_launch(void* const* d_in, const int* in_sizes, int n_in,
                              void* d_out, int out_size, void* d_ws, size_t ws_size,
                              hipStream_t stream) {
  (void)in_sizes; (void)n_in; (void)out_size; (void)ws_size;
  const float* x     = (const float*)d_in[0];
  const float* Wg    = (const float*)d_in[1];
  const float* W1    = (const float*)d_in[2];
  const float* W2    = (const float*)d_in[3];
  const float* alpha = (const float*)d_in[4];
  float* out = (float*)d_out;

  char* ws = (char*)d_ws;
  size_t o = 0;
  auto take = [&](size_t bytes) -> void* {
    void* p = ws + o;
    o = (o + bytes + 255) & ~(size_t)255;
    return p;
  };
  int*      counts   = (int*)take(Eq * sizeof(int));
  int*      offsets  = (int*)take(Eq * sizeof(int));
  int*      cursor   = (int*)take(Eq * sizeof(int));
  int*      topidx   = (int*)take((size_t)NROWS * sizeof(int));
  float*    topscale = (float*)take((size_t)NROWS * sizeof(float));
  int*      rowmap   = (int*)take((size_t)NROWS * sizeof(int));
  float*    rowscale = (float*)take((size_t)NROWS * sizeof(float));
  int*      tokpos   = (int*)take((size_t)NROWS * sizeof(int));
  _Float16* xh       = (_Float16*)take((size_t)Bq * Hq * sizeof(_Float16));     // 4 MB
  _Float16* hbuf     = (_Float16*)take((size_t)NROWS * Fq * sizeof(_Float16));  // 32 MB
  float*    eobuf    = (float*)take((size_t)NROWS * Hq * sizeof(float));        // 16 MB

  (void)hipMemsetAsync(counts, 0, Eq * sizeof(int), stream);

  cvt_x_kernel   <<<(Bq * Hq / 4) / 256, 256, 0, stream>>>(x, xh);
  gate_kernel    <<<(Bq + 255) / 256,    256, 0, stream>>>(x, Wg, alpha, topidx, topscale, counts);
  offsets_kernel <<<1, 1, 0, stream>>>(counts, offsets, cursor);
  dispatch_kernel<<<(Bq + 255) / 256,    256, 0, stream>>>(topidx, topscale, cursor, rowmap, rowscale, tokpos);
  gemm1_kernel   <<<Eq * (Bq / BT) * (Fq / BT), 256, 0, stream>>>(xh, W1, counts, offsets, rowmap, hbuf);
  gemm2_kernel   <<<Eq * (Bq / BT) * (Hq / BT), 256, 0, stream>>>(hbuf, W2, counts, offsets, rowscale, eobuf);
  combine_kernel <<<(Bq * Hq / 4) / 256, 256, 0, stream>>>(eobuf, tokpos, out);
}
